// InputAttention_44152263803489
// MI455X (gfx1250) — matrix operand order repaired and checked by reading, not compiled, not run on hardware
//
#include <hip/hip_runtime.h>
#include <hip/hip_bf16.h>
#include <math.h>

// ---------- types ----------
typedef __attribute__((ext_vector_type(16))) __bf16 v16bf;
typedef __attribute__((ext_vector_type(8)))  float  v8f;
typedef __attribute__((ext_vector_type(4)))  float  fvec4;
typedef __attribute__((ext_vector_type(4)))  unsigned int uivec4;

union FragBF {
    uivec4 q[2];
    v16bf  v;
};

static __device__ __forceinline__ unsigned short f2bf(float f) {
    unsigned u = __float_as_uint(f);
    unsigned r = (u + 0x7FFFu + ((u >> 16) & 1u)) >> 16;   // round-to-nearest-even
    return (unsigned short)r;
}

#define B_  256
#define T_  128
#define F_  1024
#define H_  512
#define IN1 1152   // T + 2H

// ---------- stage 0: W1[:, :T] -> bf16 (B-matrix source, row-major [t][k]) ----------
__global__ __launch_bounds__(256) void prep_w1(const float* __restrict__ W1,
                                               unsigned short* __restrict__ w1b) {
    int i = blockIdx.x * 256 + threadIdx.x;      // 16384 total
    int t = i >> 7, k = i & 127;
    w1b[i] = f2bf(W1[t * IN1 + k]);
}

// ---------- stage 1: base[b,t] = b1[t] + h[b]·W1[t,T:T+H] + s[b]·W1[t,T+H:] ----------
__global__ __launch_bounds__(128) void base_kernel(const float* __restrict__ h,
                                                   const float* __restrict__ s,
                                                   const float* __restrict__ W1,
                                                   const float* __restrict__ b1,
                                                   float* __restrict__ baseBT) {
    int b = blockIdx.x;
    int t = threadIdx.x;                          // 0..127
    const fvec4* h4 = (const fvec4*)(h + b * H_);
    const fvec4* s4 = (const fvec4*)(s + b * H_);
    const fvec4* wh = (const fvec4*)(W1 + t * IN1 + T_);
    const fvec4* ws = (const fvec4*)(W1 + t * IN1 + T_ + H_);
    float acc = b1[t];
    #pragma unroll 4
    for (int j = 0; j < H_ / 4; ++j) {
        fvec4 hv = h4[j], wv = wh[j];
        acc += hv[0]*wv[0] + hv[1]*wv[1] + hv[2]*wv[2] + hv[3]*wv[3];
        fvec4 sv = s4[j], w2v = ws[j];
        acc += sv[0]*w2v[0] + sv[1]*w2v[1] + sv[2]*w2v[2] + sv[3]*w2v[3];
    }
    baseBT[b * T_ + t] = acc;
}

// ---------- stage 2: per-b GEMM [64f x 128k]x[128k x 128t] via WMMA bf16, fused tanh·W2 ----------
#define BM 64
__global__ __launch_bounds__(128) void attn_main(const float* __restrict__ x,
                                                 const float* __restrict__ W2,
                                                 const unsigned short* __restrict__ w1b,
                                                 const float* __restrict__ baseBT,
                                                 float* __restrict__ eBF) {
    __shared__ __align__(16) unsigned short ldsA[BM * 136];   // [f][k], padded stride
    __shared__ __align__(16) unsigned short ldsB[T_ * 136];   // [t][k], padded stride
    __shared__ float ldsBase[T_];
    __shared__ float ldsW2[T_];

    const int b  = blockIdx.y;
    const int f0 = blockIdx.x * BM;
    const int tid = threadIdx.x;

    // --- load & convert A tile: x[b, k, f0+f] -> bf16 ldsA[f][k] ---
    const float* xb = x + (size_t)b * T_ * F_;
    #pragma unroll
    for (int i = 0; i < 16; ++i) {
        int idx = i * 128 + tid;                  // 2048 float4 chunks
        int k   = idx >> 4;                       // 16 float4 per k-row
        int f4  = idx & 15;
        fvec4 v = *(const fvec4*)(xb + k * F_ + f0 + f4 * 4);
        #pragma unroll
        for (int c = 0; c < 4; ++c)
            ldsA[(f4 * 4 + c) * 136 + k] = f2bf(v[c]);
    }
    // --- load B tile (already bf16, row-major [t][k]) ---
    #pragma unroll
    for (int i = 0; i < 16; ++i) {
        int idx = i * 128 + tid;                  // 2048 8-half octets
        int t   = idx >> 4;
        int ko  = idx & 15;
        uivec4 v = *(const uivec4*)(w1b + t * 128 + ko * 8);
        *(uivec4*)(ldsB + t * 136 + ko * 8) = v;
    }
    ldsBase[tid] = baseBT[b * T_ + tid];
    ldsW2[tid]   = W2[tid];
    __syncthreads();

    const int lane = tid & 31;
    const int wave = tid >> 5;                    // 0..3, owns f rows wave*16..wave*16+15
    const int half = lane >> 4;                   // 0 | 1
    const int col  = lane & 15;
    const int m    = wave * 16 + col;             // A row (f within tile)

    v8f acc[8];
    #pragma unroll
    for (int n = 0; n < 8; ++n) acc[n] = (v8f){0.f,0.f,0.f,0.f,0.f,0.f,0.f,0.f};

    #pragma unroll
    for (int kk = 0; kk < 4; ++kk) {
        const int k0 = kk * 32;
        // A fragment: lanes 0-15 -> K k0+0..7 | k0+16..23 ; lanes 16-31 -> +8
        FragBF fa;
        const int khA = half * 8;
        fa.q[0] = *(const uivec4*)(ldsA + m * 136 + k0 + khA);
        fa.q[1] = *(const uivec4*)(ldsA + m * 136 + k0 + 16 + khA);
        #pragma unroll
        for (int n = 0; n < 8; ++n) {
            const int t = n * 16 + col;           // B column
            FragBF fb;                            // same K order as A: k0+0..7 | k0+16..23, +8 for lanes 16-31
            const int khB = half * 8;
            fb.q[0] = *(const uivec4*)(ldsB + t * 136 + k0 + khB);
            fb.q[1] = *(const uivec4*)(ldsB + t * 136 + k0 + 16 + khB);
            acc[n] = __builtin_amdgcn_wmma_f32_16x16x32_bf16(
                         false, fa.v, false, fb.v, (short)0, acc[n], false, false);
        }
    }

    // --- fused epilogue: e[f] = sum_t tanh(u[f,t] + base[t]) * w2[t] ---
    float p[8];
    #pragma unroll
    for (int j = 0; j < 8; ++j) p[j] = 0.f;
    #pragma unroll
    for (int n = 0; n < 8; ++n) {
        const int t  = n * 16 + col;
        const float bs = ldsBase[t];
        const float wv = ldsW2[t];
        #pragma unroll
        for (int j = 0; j < 8; ++j)
            p[j] += tanhf(acc[n][j] + bs) * wv;   // lane holds u[f=wave*16+j+8*half, t]
    }
    #pragma unroll
    for (int j = 0; j < 8; ++j) {
        float v = p[j];                           // reduce over 16 lanes of this half
        #pragma unroll
        for (int off = 1; off < 16; off <<= 1) v += __shfl_xor(v, off, 32);
        if (col == 0) {
            int f = f0 + wave * 16 + j + half * 8;
            eBF[b * F_ + f] = v;
        }
    }
}

// ---------- stage 3: softmax over f per b ----------
__global__ __launch_bounds__(256) void softmax_kernel(const float* __restrict__ eBF,
                                                      float* __restrict__ out) {
    __shared__ float red[256];
    const int b = blockIdx.x, tid = threadIdx.x;
    float vals[4];
    float mx = -INFINITY;
    #pragma unroll
    for (int i = 0; i < 4; ++i) {
        vals[i] = eBF[b * F_ + tid + i * 256];
        mx = fmaxf(mx, vals[i]);
    }
    red[tid] = mx; __syncthreads();
    for (int s = 128; s > 0; s >>= 1) {
        if (tid < s) red[tid] = fmaxf(red[tid], red[tid + s]);
        __syncthreads();
    }
    mx = red[0]; __syncthreads();
    float sum = 0.f;
    #pragma unroll
    for (int i = 0; i < 4; ++i) { vals[i] = expf(vals[i] - mx); sum += vals[i]; }
    red[tid] = sum; __syncthreads();
    for (int s = 128; s > 0; s >>= 1) {
        if (tid < s) red[tid] += red[tid + s];
        __syncthreads();
    }
    const float inv = 1.f / red[0];
    #pragma unroll
    for (int i = 0; i < 4; ++i)
        out[b * F_ + tid + i * 256] = vals[i] * inv;
}

extern "C" void kernel_launch(void* const* d_in, const int* in_sizes, int n_in,
                              void* d_out, int out_size, void* d_ws, size_t ws_size,
                              hipStream_t stream) {
    (void)in_sizes; (void)n_in; (void)out_size; (void)ws_size;
    const float* h  = (const float*)d_in[0];
    const float* s  = (const float*)d_in[1];
    const float* x  = (const float*)d_in[2];
    const float* W1 = (const float*)d_in[3];
    const float* b1 = (const float*)d_in[4];
    const float* W2 = (const float*)d_in[5];
    // d_in[6] (b2) is a per-b constant shift -> softmax-invariant, skipped.

    char* ws = (char*)d_ws;
    unsigned short* w1b  = (unsigned short*)ws;                 //  32768 B
    float*          baseBT = (float*)(ws + 32768);              // 131072 B
    float*          eBF    = (float*)(ws + 32768 + 131072);     // 1 MB

    prep_w1      <<<64,  256, 0, stream>>>(W1, w1b);
    base_kernel  <<<B_,  128, 0, stream>>>(h, s, W1, b1, baseBT);
    attn_main    <<<dim3(F_ / BM, B_), 128, 0, stream>>>(x, W2, w1b, baseBT, eBF);
    softmax_kernel<<<B_, 256, 0, stream>>>(eBF, (float*)d_out);
}
